// FeatureAdaption_23244363006287
// MI455X (gfx1250) — matrix operand order repaired and checked by reading, not compiled, not run on hardware
//
#include <hip/hip_runtime.h>
#include <hip/hip_bf16.h>

typedef __attribute__((ext_vector_type(16))) __bf16 v16bf;
typedef __attribute__((ext_vector_type(8)))  float  v8f;

#define HW_    16384      // 128*128
#define CIN_   256
#define COUT_  256
#define KTOT_  2304       // DG(4) * cg(64) * K2(9)
#define NCHUNK 72         // KTOT_/32

// ---------------------------------------------------------------------------
// Kernel 1: convert w_adapt (f32 [256,256,3,3]) to bf16 in d_ws, pre-swizzled
// into the WMMA A-matrix per-lane register layout, K permuted as
// kperm = g*576 + k2*64 + c  so one 32-wide chunk = fixed (g,k2), c0..c0+31.
// Element gid = ((kc*16 + mt)*32 + lane)*16 + i
// A layout (16-bit A 16x32): K = (i<8 ? i : i+8) + (lane>=16 ? 8 : 0)
// ---------------------------------------------------------------------------
__global__ __launch_bounds__(256) void fa_prep_weights(
    const float* __restrict__ w_adapt, __bf16* __restrict__ wsA) {
  int gid = blockIdx.x * 256 + threadIdx.x;
  if (gid >= NCHUNK * 16 * 32 * 16) return;
  int i    = gid & 15;
  int lane = (gid >> 4) & 31;
  int mt   = (gid >> 9) & 15;
  int kc   = gid >> 13;

  int cout = mt * 16 + (lane & 15);
  int K    = (i < 8 ? i : i + 8) + (lane >> 4) * 8;   // 0..31 within chunk
  int g    = kc / 18;
  int t2   = kc % 18;
  int k2   = t2 >> 1;
  int c    = (t2 & 1) * 32 + K;
  int cin  = g * 64 + c;
  float v = w_adapt[(size_t)cout * KTOT_ + cin * 9 + k2];
  wsA[gid] = (__bf16)v;
}

// ---------------------------------------------------------------------------
// Kernel 2: fused offset-projection + deformable bilinear gather + implicit
// GEMM via v_wmma_f32_16x16x32_bf16 + ReLU.
// One block = 16 output pixels x 256 cout. 256 threads = 8 waves,
// each wave owns two 16-row cout tiles. B panel is double-buffered (one
// barrier per K-chunk). Corner validity folded into bilinear weights
// (branchless gathers); corner displacements stored as BYTE offsets so the
// gathers use block-uniform-base + 32-bit VGPR offset addressing (GVS mode,
// one 32-bit add per corner instead of 64-bit address chains).
// ---------------------------------------------------------------------------
__global__ __launch_bounds__(256) void fa_deform_conv(
    const float* __restrict__ x,        // [2,256,128,128]
    const float* __restrict__ shape,    // [2,2,128,128]
    const float* __restrict__ w_offset, // [72,2]
    const __bf16* __restrict__ wsA,     // preprocessed weights
    float* __restrict__ out)            // [2,256,128,128]
{
  __shared__ v16bf  panelV[2][32];  // ping-pong 32x16 bf16 B panels
  __shared__ int4   s_off[576];     // 4 clamped corner BYTE offsets (pair, px)
  __shared__ float4 s_wt[576];      // 4 masked bilinear weights   (pair, px)

  const int tid  = threadIdx.x;
  const int lane = tid & 31;
  const int wave = tid >> 5;

  const int p0    = blockIdx.x << 4;       // first flat pixel of tile
  const int n     = p0 >> 14;              // / HW
  const int h     = (p0 >> 7) & 127;
  const int wbase = p0 & 127;              // W=128 divisible by 16: one row

  // Block-uniform base of image n (scalar 64-bit); per-lane byte offsets only.
  const char* __restrict__ xbase =
      (const char*)(x + ((size_t)n * CIN_ << 14));

  // ---- precompute bilinear corner offsets/weights for all (px, g, k2) ----
  for (int item = tid; item < 576; item += 256) {
    int px   = item & 15;
    int pair = item >> 4;                  // g*9 + k2
    int k2   = pair % 9;
    int w    = wbase + px;
    float s0 = shape[((size_t)n * 2 + 0) * HW_ + (h << 7) + w];
    float s1 = shape[((size_t)n * 2 + 1) * HW_ + (h << 7) + w];
    int o = pair << 1;                     // dy channel; dx = o+1
    float dy = w_offset[2 * o    ] * s0 + w_offset[2 * o + 1] * s1;
    float dx = w_offset[2 * o + 2] * s0 + w_offset[2 * o + 3] * s1;
    float py  = dy + (float)(h + k2 / 3 - 1);
    float pxf = dx + (float)(w + k2 % 3 - 1);
    float fy = floorf(py), fx = floorf(pxf);
    int y0 = (int)fy,      x0 = (int)fx;
    float ly = py - fy,    lx = pxf - fx;
    float hy = 1.f - ly,   hx = 1.f - lx;
    int y1 = y0 + 1, x1 = x0 + 1;
    float vy0 = ((unsigned)y0 < 128u) ? 1.f : 0.f;
    float vy1 = ((unsigned)y1 < 128u) ? 1.f : 0.f;
    float vx0 = ((unsigned)x0 < 128u) ? 1.f : 0.f;
    float vx1 = ((unsigned)x1 < 128u) ? 1.f : 0.f;
    int yc0 = min(max(y0, 0), 127), yc1 = min(max(y1, 0), 127);
    int xc0 = min(max(x0, 0), 127), xc1 = min(max(x1, 0), 127);
    // byte offsets within one channel plane (<< 2 for f32)
    s_off[item] = make_int4((((yc0 << 7) + xc0) << 2), (((yc0 << 7) + xc1) << 2),
                            (((yc1 << 7) + xc0) << 2), (((yc1 << 7) + xc1) << 2));
    s_wt[item]  = make_float4(hy * hx * vy0 * vx0, hy * lx * vy0 * vx1,
                              ly * hx * vy1 * vx0, ly * lx * vy1 * vx1);
  }
  __syncthreads();

  v8f acc0{}; v8f acc1{};
  const int mt0 = wave * 2, mt1 = mt0 + 1;
  const v16bf* wsV = (const v16bf*)wsA;

  for (int kc = 0; kc < NCHUNK; ++kc) {
    int g  = kc / 18;
    int t2 = kc % 18;
    int k2 = t2 >> 1;
    int pairBase = (g * 9 + k2) << 4;
    int cinBase  = g * 64 + (t2 & 1) * 32;
    __bf16* panel = (__bf16*)panelV[kc & 1];

    if (kc + 1 < NCHUNK) {  // warm caches for next chunk's A fragments
      __builtin_prefetch(&wsA[(((size_t)(kc + 1) * 16 + mt0) * 32 + lane) * 16], 0, 1);
      __builtin_prefetch(&wsA[(((size_t)(kc + 1) * 16 + mt1) * 32 + lane) * 16], 0, 1);
    }

    // ---- gather 32 channels x 16 pixels into bf16 B panel (branchless) ---
    int item = tid;
#pragma unroll
    for (int rep = 0; rep < 2; ++rep, item += 256) {
      int px  = item & 15;
      int k   = item >> 4;                 // 0..31 -> channel cinBase+k
      int idx = pairBase + px;
      int4   o  = s_off[idx];              // one ds_load_b128
      float4 wt = s_wt[idx];               // one ds_load_b128
      int chan  = (cinBase + k) << 16;     // channel-plane byte offset (HW*4)
      // block-uniform base + 32-bit per-lane byte offsets -> GVS addressing
      float a00 = *(const float*)(xbase + (chan + o.x));
      float a01 = *(const float*)(xbase + (chan + o.y));
      float a10 = *(const float*)(xbase + (chan + o.z));
      float a11 = *(const float*)(xbase + (chan + o.w));
      float v = wt.x * a00 + wt.y * a01 + wt.z * a10 + wt.w * a11;
      // B 32x16 layout: lanes 0-15 hold K=0..7,16..23, lanes 16-31 hold K=8..15,24..31
      panel[((px + (((k >> 3) & 1) << 4)) << 4) + ((k & 7) + ((k >> 4) << 3))] = (__bf16)v;
    }
    __syncthreads();   // panel (kc&1) filled; prior-iteration reads of the
                       // other buffer were ordered by the previous barrier

    v16bf b  = panelV[kc & 1][lane];                           // ds_load_b128 x2
    v16bf a0 = wsV[((size_t)kc * 16 + mt0) * 32 + lane];       // global_load_b128 x2
    v16bf a1 = wsV[((size_t)kc * 16 + mt1) * 32 + lane];
    acc0 = __builtin_amdgcn_wmma_f32_16x16x32_bf16(false, a0, false, b,
                                                   (short)0, acc0, false, false);
    acc1 = __builtin_amdgcn_wmma_f32_16x16x32_bf16(false, a1, false, b,
                                                   (short)0, acc1, false, false);
    // no second barrier: next iteration writes the other panel buffer
  }

  // ---- writeback with ReLU; C layout: VGPR r -> M row, lane -> N column --
  int pxcol = lane & 15;
  int mrow  = (lane >> 4) << 3;            // lanes>=16 hold rows 8..15
  int w = wbase + pxcol;
  size_t outBase = (size_t)n * COUT_ * HW_ + (h << 7) + w;
#pragma unroll
  for (int r = 0; r < 8; ++r) {
    int c0 = mt0 * 16 + mrow + r;
    int c1 = mt1 * 16 + mrow + r;
    out[outBase + (size_t)c0 * HW_] = fmaxf(acc0[r], 0.f);
    out[outBase + (size_t)c1 * HW_] = fmaxf(acc1[r], 0.f);
  }
}

// ---------------------------------------------------------------------------
extern "C" void kernel_launch(void* const* d_in, const int* in_sizes, int n_in,
                              void* d_out, int out_size, void* d_ws, size_t ws_size,
                              hipStream_t stream) {
  const float* x        = (const float*)d_in[0];  // [2,256,128,128]
  const float* shape    = (const float*)d_in[1];  // [2,2,128,128]
  const float* w_offset = (const float*)d_in[2];  // [72,2]
  const float* w_adapt  = (const float*)d_in[3];  // [256,256,3,3]
  float*       out      = (float*)d_out;          // [2,256,128,128]
  __bf16*      wsA      = (__bf16*)d_ws;          // 589824 bf16 = 1.125 MiB

  // weights -> bf16, WMMA A-layout swizzle (recomputed every call: stateless)
  fa_prep_weights<<<2304, 256, 0, stream>>>(w_adapt, wsA);
  // fused offsets + deformable gather + implicit GEMM + ReLU
  fa_deform_conv<<<2048, 256, 0, stream>>>(x, shape, w_offset, wsA, out);
}
